// fMRI3DGNN_16595753631839
// MI455X (gfx1250) — matrix operand order repaired and checked by reading, not compiled, not run on hardware
//
#include <hip/hip_runtime.h>
#include <hip/hip_bf16.h>
#include <math.h>

typedef __attribute__((ext_vector_type(16))) _Float16 v16h;
typedef __attribute__((ext_vector_type(8)))  float    v8f;
typedef __attribute__((ext_vector_type(4)))  float    v4f;

#define NB    16
#define NROI  100
#define FLATN 10000   // NROI*NROI

__device__ __forceinline__ float gelu_exact(float x) {
    return 0.5f * x * (1.0f + erff(x * 0.7071067811865475f));
}
__device__ __forceinline__ float wave_sum(float v) {
#pragma unroll
    for (int off = 16; off > 0; off >>= 1) v += __shfl_xor(v, off, 32);
    return v;
}
__device__ __forceinline__ float wave_max(float v) {
#pragma unroll
    for (int off = 16; off > 0; off >>= 1) v = fmaxf(v, __shfl_xor(v, off, 32));
    return v;
}

// ---------------------------------------------------------------------------
// Generic GEMM: C[M,N] = act(A[M,K] @ B[K,N] + bias[N])
// fp32 in memory, converted to f16 in registers, f32 accumulate via
// v_wmma_f32_16x16x32_f16. One wave owns one 16x16 output tile.
// N is a template constant so all strided-B offsets ((h+(h&8))*N*4 <= 920KB, within
// the signed 24-bit instruction offset) fold into immediate offsets, and the
// hot loop is pointer-bump only. A loads are 4x b128 per K-step.
// M, N multiples of 16 (true at all call sites); K guarded via tail.
// act: 0 = identity, 1 = sigmoid.
// ---------------------------------------------------------------------------
template <int N>
__global__ __launch_bounds__(256) void gemm_wmma_f16(
    const float* __restrict__ A, const float* __restrict__ B,
    const float* __restrict__ bias, float* __restrict__ C,
    int M, int K, int act)
{
    const int lane = threadIdx.x & 31;
    const int wave = blockIdx.x * (blockDim.x >> 5) + (threadIdx.x >> 5);
    const int ntiles = N >> 4;
    const int mtiles = M >> 4;
    if (wave >= ntiles * mtiles) return;          // wave-uniform: EXEC stays full
    const int mt = wave / ntiles;
    const int nt = wave % ntiles;

    // A fragment (16x32 f16): lanes 0-15 -> M=lane, K in {0..7,16..23};
    //                         lanes 16-31 -> M=lane-16, K in {8..15,24..31}
    const int arow = mt * 16 + (lane & 15);
    const int klo  = (lane < 16) ? 0 : 8;
    // B fragment (32x16 f16): lane%16 = N col; lanes<16 hold K {0..7,16..23}, else {8..15,24..31}
    const int kbo  = (lane < 16) ? 0 : 8;
    const int ncol = nt * 16 + (lane & 15);
    const float* Arow = A + (size_t)arow * K;

    const float* Ap = Arow + klo;                     // per-lane A base
    const float* Bp = B + (size_t)kbo * N + ncol;     // per-lane B base

    v8f acc = {};
    const int kIter = K >> 5;
    for (int it = 0; it < kIter; ++it, Ap += 32, Bp += (size_t)32 * N) {
        // A: 4x contiguous float4 (16B-aligned: bases/strides are 16B multiples)
        v4f a0 = *(const v4f*)(Ap);
        v4f a1 = *(const v4f*)(Ap + 4);
        v4f a2 = *(const v4f*)(Ap + 16);
        v4f a3 = *(const v4f*)(Ap + 20);
        v16h a, b;
#pragma unroll
        for (int t = 0; t < 4; ++t) {
            a[t]      = (_Float16)a0[t];
            a[t + 4]  = (_Float16)a1[t];
            a[t + 8]  = (_Float16)a2[t];
            a[t + 12] = (_Float16)a3[t];
        }
#pragma unroll
        for (int h = 0; h < 16; ++h)
            b[h] = (_Float16)Bp[(size_t)(h + (h & 8)) * N];       // imm-offset strided loads
        acc = __builtin_amdgcn_wmma_f32_16x16x32_f16(false, a, false, b,
                                                     (short)0, acc, false, false);
    }
    const int k0 = kIter << 5;
    if (k0 < K) {                                     // zero-padded K tail
        v16h a, b;
#pragma unroll
        for (int t = 0; t < 8; ++t) {
            int k1 = k0 + klo + t, k2 = k0 + 16 + klo + t;
            a[t]     = (k1 < K) ? (_Float16)Arow[k1] : (_Float16)0.0f;
            a[t + 8] = (k2 < K) ? (_Float16)Arow[k2] : (_Float16)0.0f;
        }
        const float* Bt = B + ncol;
#pragma unroll
        for (int h = 0; h < 16; ++h) {
            int kk = k0 + kbo + h + (h & 8);
            b[h] = (kk < K) ? (_Float16)Bt[(size_t)kk * N] : (_Float16)0.0f;
        }
        acc = __builtin_amdgcn_wmma_f32_16x16x32_f16(false, a, false, b,
                                                     (short)0, acc, false, false);
    }
    // C/D layout: VGPR v -> M = v + 8*(lane>=16), N = lane%16
    const float bv = bias ? bias[ncol] : 0.0f;
    const int rbase = mt * 16 + ((lane < 16) ? 0 : 8);
#pragma unroll
    for (int v = 0; v < 8; ++v) {
        float val = acc[v] + bv;
        if (act == 1) val = 1.0f / (1.0f + expf(-val));
        C[(size_t)(rbase + v) * N + ncol] = val;
    }
}

// adj = (P + P^T) * 0.5 per batch
__global__ void symmetrize_kernel(const float* __restrict__ P, float* __restrict__ S) {
    int idx = blockIdx.x * blockDim.x + threadIdx.x;
    if (idx >= NB * FLATN) return;
    int b = idx / FLATN, r = idx % FLATN;
    int i = r / NROI, j = r % NROI;
    S[idx] = 0.5f * (P[idx] + P[b * FLATN + j * NROI + i]);
}

// Exact 0.75 quantile (linear interp): thr = v[7499] + 0.25*(v[7500]-v[7499]).
// Order statistics via bit-level binary search (sigmoid outputs are positive
// floats -> monotone as uint32). One block per batch.
__global__ __launch_bounds__(1024) void quantile_kernel(
    const float* __restrict__ S, float* __restrict__ thr)
{
    const int b = blockIdx.x;
    const float* d = S + (size_t)b * FLATN;
    __shared__ int cnt;
    unsigned kth[2];
    const int ks[2] = {7500, 7501};                // 1-based ranks of v[7499], v[7500]
    for (int q = 0; q < 2; ++q) {
        unsigned lo = 0u, hi = 0x3F800000u;        // (0, 1.0f]
        while (lo < hi) {
            unsigned mid = (lo + hi) >> 1u;
            if (threadIdx.x == 0) cnt = 0;
            __syncthreads();
            int c = 0;
            for (int t = threadIdx.x; t < FLATN; t += 1024)
                if (__float_as_uint(d[t]) <= mid) ++c;
            atomicAdd(&cnt, c);
            __syncthreads();
            int total = cnt;
            __syncthreads();
            if (total >= ks[q]) hi = mid; else lo = mid + 1u;
        }
        kth[q] = lo;
    }
    if (threadIdx.x == 0) {
        float v0 = __uint_as_float(kth[0]);
        float v1 = __uint_as_float(kth[1]);
        thr[b] = v0 + 0.25f * (v1 - v0);
    }
}

// per-row mean and unbiased std -> b_ft[B*N, 2]; one wave per row
__global__ __launch_bounds__(256) void rowstats_kernel(
    const float* __restrict__ S, float* __restrict__ bft)
{
    int wave = blockIdx.x * 8 + (threadIdx.x >> 5);
    int lane = threadIdx.x & 31;
    if (wave >= NB * NROI) return;
    const float* row = S + (size_t)wave * NROI;
    float s1 = 0.f, s2 = 0.f;
#pragma unroll
    for (int t = 0; t < 4; ++t) {
        int j = lane + 32 * t;
        if (j < NROI) { float v = row[j]; s1 += v; s2 += v * v; }
    }
    s1 = wave_sum(s1); s2 = wave_sum(s2);
    if (lane == 0) {
        float mean = s1 * (1.0f / NROI);
        float var  = (s2 - NROI * mean * mean) * (1.0f / (NROI - 1));
        bft[wave * 2 + 0] = mean;
        bft[wave * 2 + 1] = sqrtf(fmaxf(var, 0.0f)) + 1e-6f;
    }
}

// Linear(2,8)->GELU->Linear(8,16)->LayerNorm(16); one thread per node
__global__ void featenh_kernel(const float* __restrict__ bft,
                               const float* __restrict__ W1, const float* __restrict__ b1,
                               const float* __restrict__ W2, const float* __restrict__ b2,
                               const float* __restrict__ g,  const float* __restrict__ beta,
                               float* __restrict__ x0)
{
    int idx = blockIdx.x * blockDim.x + threadIdx.x;
    if (idx >= NB * NROI) return;
    float m = bft[idx * 2], s = bft[idx * 2 + 1];
    float hid[8];
#pragma unroll
    for (int u = 0; u < 8; ++u)
        hid[u] = gelu_exact(m * W1[u] + s * W1[8 + u] + b1[u]);
    float o[16]; float mu = 0.f;
#pragma unroll
    for (int v = 0; v < 16; ++v) {
        float t = b2[v];
#pragma unroll
        for (int u = 0; u < 8; ++u) t += hid[u] * W2[u * 16 + v];
        o[v] = t; mu += t;
    }
    mu *= (1.0f / 16.0f);
    float var = 0.f;
#pragma unroll
    for (int v = 0; v < 16; ++v) { float d = o[v] - mu; var += d * d; }
    var *= (1.0f / 16.0f);
    float inv = rsqrtf(var + 1e-5f);
#pragma unroll
    for (int v = 0; v < 16; ++v)
        x0[idx * 16 + v] = (o[v] - mu) * inv * g[v] + beta[v];
}

// ---------------------------------------------------------------------------
// GATv2 attention + aggregation + bias + GELU. One wave per (b, i, h):
// lanes span channels (CR = C/32 regs/lane); 100 logits kept 4-per-lane.
// e[j] = att_h . leakyrelu(xl[j]+xr[i]) if edge else -1e9; softmax over j;
// alpha zeroed on non-edges; out = sum_j alpha_j * xl[j]; then gelu(out+bias).
// ---------------------------------------------------------------------------
template <int CR>
__global__ __launch_bounds__(256) void gat_attn_kernel(
    const float* __restrict__ xl, const float* __restrict__ xr,
    const float* __restrict__ att, const float* __restrict__ bias,
    const float* __restrict__ adj, const float* __restrict__ thr,
    float* __restrict__ out, int H, int selfloop)
{
    const int C = CR * 32;
    const int lane = threadIdx.x & 31;
    int wave = blockIdx.x * 8 + (threadIdx.x >> 5);
    if (wave >= NB * NROI * H) return;             // wave-uniform exit
    int b = wave / (NROI * H);
    int r = wave % (NROI * H);
    int i = r / H;
    int h = r % H;
    const int HC = H * C;

    const float* xrp  = xr  + (size_t)(b * NROI + i) * HC + h * C;
    const float* attp = att + h * C;
    float xri[CR], av[CR];
#pragma unroll
    for (int c = 0; c < CR; ++c) {
        xri[c] = xrp[lane + 32 * c];
        av[c]  = attp[lane + 32 * c];
    }
    const float tv = thr[b];
    const float* adjrow = adj + (size_t)b * FLATN + i * NROI;
    const float* xlb = xl + (size_t)(b * NROI) * HC + h * C;

    float el[4];
#pragma unroll
    for (int t = 0; t < 4; ++t) {
        el[t] = -INFINITY;
        const int jmax = (t < 3) ? 32 : (NROI - 96);
        for (int jj = 0; jj < jmax; ++jj) {
            int j = t * 32 + jj;
            bool mk = (adjrow[j] > tv) || (selfloop && (i == j));   // uniform
            float e;
            if (mk) {
                const float* xlj = xlb + (size_t)j * HC;
                float part = 0.f;
#pragma unroll
                for (int c = 0; c < CR; ++c) {
                    float v = xlj[lane + 32 * c] + xri[c];
                    v = (v > 0.f) ? v : 0.2f * v;
                    part += av[c] * v;
                }
                e = wave_sum(part);
            } else {
                e = -1e9f;
            }
            el[t] = (jj == lane) ? e : el[t];
        }
    }
    float mx = fmaxf(fmaxf(el[0], el[1]), fmaxf(el[2], el[3]));
    mx = wave_max(mx);
    float al[4]; float ssum = 0.f;
#pragma unroll
    for (int t = 0; t < 4; ++t) { al[t] = expf(el[t] - mx); ssum += al[t]; }
    ssum = wave_sum(ssum);
    const float inv = 1.0f / ssum;

    float accv[CR];
#pragma unroll
    for (int c = 0; c < CR; ++c) accv[c] = 0.f;
#pragma unroll
    for (int t = 0; t < 4; ++t) {
        const int jmax = (t < 3) ? 32 : (NROI - 96);
        for (int jj = 0; jj < jmax; ++jj) {
            int j = t * 32 + jj;
            bool mk = (adjrow[j] > tv) || (selfloop && (i == j));   // uniform
            if (!mk) continue;                      // alpha -> 0 on non-edges
            float a = __shfl(al[t], jj, 32) * inv;
            const float* xlj = xlb + (size_t)j * HC;
#pragma unroll
            for (int c = 0; c < CR; ++c) accv[c] += a * xlj[lane + 32 * c];
        }
    }
    float* op = out + (size_t)(b * NROI + i) * HC + h * C;
    const float* bp = bias + h * C;
#pragma unroll
    for (int c = 0; c < CR; ++c)
        op[lane + 32 * c] = gelu_exact(accv[c] + bp[lane + 32 * c]);
}

// global mean pool over nodes: out[b,c] = mean_i x3[b,i,c]
__global__ void pool_kernel(const float* __restrict__ x3, float* __restrict__ outp) {
    int idx = blockIdx.x * blockDim.x + threadIdx.x;
    if (idx >= NB * 128) return;
    int b = idx / 128, c = idx % 128;
    float s = 0.f;
    for (int i = 0; i < NROI; ++i) s += x3[(size_t)(b * NROI + i) * 128 + c];
    outp[idx] = s * (1.0f / NROI);
}

static inline int gemm_blocks(int M, int N) { return ((M >> 4) * (N >> 4) + 7) / 8; }

extern "C" void kernel_launch(void* const* d_in, const int* in_sizes, int n_in,
                              void* d_out, int out_size, void* d_ws, size_t ws_size,
                              hipStream_t stream) {
    (void)in_sizes; (void)n_in; (void)out_size; (void)ws_size;
    const float* raw   = (const float*)d_in[0];
    const float* gbW   = (const float*)d_in[1];
    const float* gbb   = (const float*)d_in[2];
    const float* feW1  = (const float*)d_in[3];
    const float* feb1  = (const float*)d_in[4];
    const float* feW2  = (const float*)d_in[5];
    const float* feb2  = (const float*)d_in[6];
    const float* feg   = (const float*)d_in[7];
    const float* febet = (const float*)d_in[8];
    const float* Wl1 = (const float*)d_in[9];   const float* bl1 = (const float*)d_in[10];
    const float* Wr1 = (const float*)d_in[11];  const float* br1 = (const float*)d_in[12];
    const float* att1= (const float*)d_in[13];  const float* bias1=(const float*)d_in[14];
    const float* Wl2 = (const float*)d_in[15];  const float* bl2 = (const float*)d_in[16];
    const float* Wr2 = (const float*)d_in[17];  const float* br2 = (const float*)d_in[18];
    const float* att2= (const float*)d_in[19];  const float* bias2=(const float*)d_in[20];
    const float* Wl3 = (const float*)d_in[21];  const float* bl3 = (const float*)d_in[22];
    const float* Wr3 = (const float*)d_in[23];  const float* br3 = (const float*)d_in[24];
    const float* att3= (const float*)d_in[25];  const float* bias3=(const float*)d_in[26];

    float* ws  = (float*)d_ws;
    float* P    = ws;                       // 16*10000
    float* adjs = P    + NB * FLATN;        // 16*10000
    float* thr  = adjs + NB * FLATN;        // 16
    float* bft  = thr  + NB;                // 1600*2
    float* x0   = bft  + NB * NROI * 2;     // 1600*16
    float* xlb  = x0   + NB * NROI * 16;    // 1600*1024
    float* xrb  = xlb  + 1600 * 1024;       // 1600*1024
    float* x1   = xrb  + 1600 * 1024;       // 1600*1024
    float* x2   = x1   + 1600 * 1024;       // 1600*1024
    float* x3   = x2   + 1600 * 1024;       // 1600*128

    // --- build_graph: sigmoid(raw @ gbW + gbb) ---
    gemm_wmma_f16<FLATN><<<gemm_blocks(NB, FLATN), 256, 0, stream>>>(
        raw, gbW, gbb, P, NB, FLATN, /*act=*/1);
    symmetrize_kernel<<<(NB * FLATN + 255) / 256, 256, 0, stream>>>(P, adjs);
    quantile_kernel<<<NB, 1024, 0, stream>>>(adjs, thr);
    rowstats_kernel<<<(NB * NROI + 7) / 8, 256, 0, stream>>>(adjs, bft);
    featenh_kernel<<<(NB * NROI + 255) / 256, 256, 0, stream>>>(
        bft, feW1, feb1, feW2, feb2, feg, febet, x0);

    // --- GAT layer 1: in=16, H=8, C=128, no self loops ---
    gemm_wmma_f16<1024><<<gemm_blocks(1600, 1024), 256, 0, stream>>>(x0, Wl1, bl1, xlb, 1600, 16, 0);
    gemm_wmma_f16<1024><<<gemm_blocks(1600, 1024), 256, 0, stream>>>(x0, Wr1, br1, xrb, 1600, 16, 0);
    gat_attn_kernel<4><<<(NB * NROI * 8 + 7) / 8, 256, 0, stream>>>(
        xlb, xrb, att1, bias1, adjs, thr, x1, 8, 0);

    // --- GAT layer 2: in=1024, H=4, C=256, self loops ---
    gemm_wmma_f16<1024><<<gemm_blocks(1600, 1024), 256, 0, stream>>>(x1, Wl2, bl2, xlb, 1600, 1024, 0);
    gemm_wmma_f16<1024><<<gemm_blocks(1600, 1024), 256, 0, stream>>>(x1, Wr2, br2, xrb, 1600, 1024, 0);
    gat_attn_kernel<8><<<(NB * NROI * 4 + 7) / 8, 256, 0, stream>>>(
        xlb, xrb, att2, bias2, adjs, thr, x2, 4, 1);

    // --- GAT layer 3: in=1024, H=1, C=128, self loops, mean==identity ---
    gemm_wmma_f16<128><<<gemm_blocks(1600, 128), 256, 0, stream>>>(x2, Wl3, bl3, xlb, 1600, 1024, 0);
    gemm_wmma_f16<128><<<gemm_blocks(1600, 128), 256, 0, stream>>>(x2, Wr3, br3, xrb, 1600, 1024, 0);
    gat_attn_kernel<4><<<(NB * NROI * 1 + 7) / 8, 256, 0, stream>>>(
        xlb, xrb, att3, bias3, adjs, thr, x3, 1, 1);

    // --- global mean pool ---
    pool_kernel<<<(NB * 128 + 255) / 256, 256, 0, stream>>>(x3, (float*)d_out);
}
